// MultiHeadAttention_35759897707215
// MI455X (gfx1250) — matrix operand order repaired and checked by reading, not compiled, not run on hardware
//
#include <hip/hip_runtime.h>
#include <hip/hip_bf16.h>
#include <stdint.h>

// ---------------------------------------------------------------------------
// Problem constants (reference: B=2, T=2048, C=1024, 16 heads, head_dim=64)
// ---------------------------------------------------------------------------
#define BB   2
#define TT   2048
#define CC   1024
#define NH   16
#define HD   64
#define MTOT (BB * TT)        // 4096 token rows
#define N3C  (3 * CC)         // 3072

typedef __attribute__((ext_vector_type(16))) __bf16 v16bf;
typedef __attribute__((ext_vector_type(8)))  float  v8f;
typedef __attribute__((ext_vector_type(4)))  unsigned int u32x4;
typedef __attribute__((ext_vector_type(8)))  int    i32x8;
typedef __attribute__((ext_vector_type(4)))  int    i32x4;

#if defined(__gfx1250__) && __has_builtin(__builtin_amdgcn_tensor_load_to_lds)
#define USE_TDM 1
#else
#define USE_TDM 0
#endif

// ---------------------------------------------------------------------------
// Helpers
// ---------------------------------------------------------------------------
static __device__ inline __bf16 tobf(float f) {
    union { float f; uint32_t u; } v; v.f = f;
    uint32_t r = v.u + 0x7FFFu + ((v.u >> 16) & 1u);   // round-to-nearest-even
    unsigned short h = (unsigned short)(r >> 16);
    return __builtin_bit_cast(__bf16, h);
}

static __device__ inline v8f wmma_bf16(v16bf a, v16bf b, v8f c) {
    // D = A(16x32 bf16) * B(32x16 bf16) + C(16x16 f32)
    return __builtin_amdgcn_wmma_f32_16x16x32_bf16(
        false, a, false, b, (short)0, c, false, false);
}

union AF { v16bf v; uint4 q[2]; };

// A-fragment (16x32, 16-bit): row M = lane&15; K chunk {hl*16..+15}
static __device__ inline v16bf load_a(const __bf16* row, int k0, int hl) {
    AF f;
    f.q[0] = *reinterpret_cast<const uint4*>(row + k0 + hl * 16);
    f.q[1] = *reinterpret_cast<const uint4*>(row + k0 + hl * 16 + 8);
    return f.v;
}

static __device__ inline float rmax16(float x) {
    x = fmaxf(x, __shfl_xor(x, 1, 16));
    x = fmaxf(x, __shfl_xor(x, 2, 16));
    x = fmaxf(x, __shfl_xor(x, 4, 16));
    x = fmaxf(x, __shfl_xor(x, 8, 16));
    return x;
}
static __device__ inline float rsum16(float x) {
    x += __shfl_xor(x, 1, 16);
    x += __shfl_xor(x, 2, 16);
    x += __shfl_xor(x, 4, 16);
    x += __shfl_xor(x, 8, 16);
    return x;
}

static __device__ inline void wave_fence() {
    __builtin_amdgcn_wave_barrier();
    asm volatile("" ::: "memory");
}

#if USE_TDM
// Build a 2-D Tensor-DMA descriptor (D#) and issue TENSOR_LOAD_TO_LDS.
// Per CDNA5 ISA 8.3/8.4: group0 = {count/flags, lds_addr, global_addr, type},
// group1 = {data_size, tensor dims, tile dims, dim0 stride}. 2-byte elements.
static __device__ inline void tdm_load_2d(uint32_t lds_off, const void* gptr,
                                          uint32_t tensor_d0, uint32_t tensor_d1,
                                          uint32_t tile_d0,   uint32_t tile_d1,
                                          uint32_t stride0) {
    uint64_t ga = (uint64_t)(uintptr_t)gptr;
    u32x4 g0;
    g0.x = 1u;                                            // count=1 (valid user D#)
    g0.y = lds_off;                                       // lds_addr (bytes)
    g0.z = (uint32_t)ga;                                  // global_addr[31:0]
    g0.w = (uint32_t)((ga >> 32) & 0x1FFFFFFu) | (2u << 30); // addr[56:32], type=2
    i32x8 g1;
    g1[0] = (int)(1u << 16);                              // data_size=1 -> 2 bytes
    g1[1] = (int)((tensor_d0 & 0xFFFFu) << 16);           // tensor_dim0[15:0]
    g1[2] = (int)(((tensor_d0 >> 16) & 0xFFFFu) |
                  ((tensor_d1 & 0xFFFFu) << 16));         // dim0 hi | dim1 lo
    g1[3] = (int)(((tensor_d1 >> 16) & 0xFFFFu) |
                  ((tile_d0 & 0xFFFFu) << 16));           // dim1 hi | tile_dim0
    g1[4] = (int)(tile_d1 & 0xFFFFu);                     // tile_dim1 (tile_dim2=0)
    g1[5] = (int)stride0;                                 // tensor_dim0_stride[31:0]
    g1[6] = 0;                                            // stride hi / dim1_stride
    g1[7] = 0;
    i32x4 z4 = {0, 0, 0, 0};
#if __clang_major__ >= 23
    i32x8 z8 = {0, 0, 0, 0, 0, 0, 0, 0};
    __builtin_amdgcn_tensor_load_to_lds(g0, g1, z4, z4, z8, 0);
#else
    __builtin_amdgcn_tensor_load_to_lds(g0, g1, z4, z4, 0);
#endif
}
#endif

// ---------------------------------------------------------------------------
// Kernel 0: fp32 -> bf16 conversions (weights transposed: B K-dim innermost)
// ---------------------------------------------------------------------------
#define NX   (MTOT * CC)
#define NWA  (CC * N3C)
#define NWP  (CC * CC)
#define NCVT (NX + NWA + NWP)

__global__ __launch_bounds__(256)
void cvt_kernel(const float* __restrict__ x,
                const float* __restrict__ Wa,
                const float* __restrict__ Wp,
                __bf16* __restrict__ xb,
                __bf16* __restrict__ WtA,   // [3C][C]
                __bf16* __restrict__ WtP)   // [C][C]
{
    for (long i = blockIdx.x * 256 + threadIdx.x; i < NCVT;
         i += (long)gridDim.x * 256) {
        if (i < NX) {
            xb[i] = tobf(x[i]);
        } else if (i < NX + NWA) {
            long k = i - NX;
            long c = k / N3C, n = k % N3C;
            WtA[n * CC + c] = tobf(Wa[k]);
        } else {
            long k = i - NX - NWA;
            long c = k >> 10, n = k & (CC - 1);
            WtP[n * CC + c] = tobf(Wp[k]);
        }
    }
}

// ---------------------------------------------------------------------------
// Register-blocked WMMA GEMM: each wave owns a 64x32 output block
// (4 M-frags x 2 N-frags), reusing A x2 and B x4 per k-step.
// A row-major [M][K]; Bt row-major [N][K].
// ---------------------------------------------------------------------------
static __device__ inline void mma_block42(const __bf16* __restrict__ A,
                                          const __bf16* __restrict__ Bt,
                                          int m0, int n0, int K,
                                          v8f (&acc)[4][2]) {
    const int lane = threadIdx.x & 31;
    const int hl = lane >> 4, l16 = lane & 15;
    const __bf16* ar[4];
    const __bf16* br[2];
    #pragma unroll
    for (int i = 0; i < 4; ++i) ar[i] = A  + (size_t)(m0 + 16 * i + l16) * K;
    #pragma unroll
    for (int j = 0; j < 2; ++j) br[j] = Bt + (size_t)(n0 + 16 * j + l16) * K;

    for (int k = 0; k < K; k += 32) {
        v16bf a[4], b[2];
        #pragma unroll
        for (int i = 0; i < 4; ++i) a[i] = load_a(ar[i], k, hl);
        #pragma unroll
        for (int j = 0; j < 2; ++j)
            b[j] = *reinterpret_cast<const v16bf*>(br[j] + k + hl * 16);
        #pragma unroll
        for (int i = 0; i < 4; ++i)
            #pragma unroll
            for (int j = 0; j < 2; ++j)
                acc[i][j] = wmma_bf16(a[i], b[j], acc[i][j]);
    }
}

// ---------------------------------------------------------------------------
// Kernel 1: QKV GEMM  (4096 x 3072; 64x32 per wave -> 64 x 96 wave-blocks)
// Epilogue scatters Q [b,h,t,d], K [b,h,t,d], V^T [b,h,d,t] as bf16.
// ---------------------------------------------------------------------------
__global__ __launch_bounds__(128)
void qkv_kernel(const __bf16* __restrict__ xb,
                const __bf16* __restrict__ WtA,
                const float*  __restrict__ b_attn,
                __bf16* __restrict__ Qb,
                __bf16* __restrict__ Kb,
                __bf16* __restrict__ Vt)
{
    const int wid = blockIdx.x * 4 + (threadIdx.x >> 5);   // < 6144
    const int mb = wid / 96, nb = wid % 96;
    const int m0 = mb * 64, n0 = nb * 32;

    v8f acc[4][2];
    #pragma unroll
    for (int i = 0; i < 4; ++i)
        #pragma unroll
        for (int j = 0; j < 2; ++j) { v8f z = {}; acc[i][j] = z; }

    mma_block42(xb, WtA, m0, n0, CC, acc);

    const int lane = threadIdx.x & 31;
    const int hl = lane >> 4, l16 = lane & 15;

    #pragma unroll
    for (int j = 0; j < 2; ++j) {
        const int n = n0 + 16 * j + l16;
        const float bv = b_attn[n];
        const int which = n >> 10;              // 0=Q 1=K 2=V
        const int col = n & (CC - 1);
        const int h = col >> 6, d = col & 63;
        #pragma unroll
        for (int i = 0; i < 4; ++i) {
            #pragma unroll
            for (int r = 0; r < 8; ++r) {
                const int m = m0 + 16 * i + r + 8 * hl;
                const int bbi = m >> 11, t = m & (TT - 1);
                const __bf16 o = tobf(acc[i][j][r] + bv);
                const size_t bh = (size_t)(bbi * NH + h);
                if (which == 0)      Qb[(bh * TT + t) * HD + d] = o;
                else if (which == 1) Kb[(bh * TT + t) * HD + d] = o;
                else                 Vt[(bh * HD + d) * TT + t] = o;
            }
        }
    }
}

// ---------------------------------------------------------------------------
// Kernel 2: flash attention. One wave = one (b, h, 16-query tile).
// K/V tiles staged LDS via Tensor Data Mover (TENSORcnt), S and P*V on WMMA,
// online softmax with width-16 shuffles, P relaid C->A through LDS.
// ---------------------------------------------------------------------------
__global__ __launch_bounds__(128)
void attn_kernel(const __bf16* __restrict__ Qb,
                 const __bf16* __restrict__ Kb,
                 const __bf16* __restrict__ Vt,
                 __bf16* __restrict__ yb)   // [4096][1024] bf16
{
    __shared__ __attribute__((aligned(16))) __bf16 Kst[4][32][HD];  // 16 KB
    __shared__ __attribute__((aligned(16))) __bf16 Vst[4][HD][32];  // 16 KB
    __shared__ __attribute__((aligned(16))) __bf16 plds[4][16][32]; //  4 KB

    const int w   = threadIdx.x >> 5;
    const int wid = blockIdx.x * 4 + w;        // < 4096
    const int qt  = wid & 127;
    const int h   = (wid >> 7) & 15;
    const int bb  = wid >> 11;

    const int lane = threadIdx.x & 31;
    const int hl = lane >> 4, l16 = lane & 15;

    const size_t bh = (size_t)(bb * NH + h);
    const __bf16* Qh = Qb + bh * TT * HD;
    const __bf16* Kh = Kb + bh * TT * HD;
    const __bf16* Vh = Vt + bh * HD * TT;

    const int q0 = qt * 16;
    const __bf16* qrow = Qh + (size_t)(q0 + l16) * HD;
    const v16bf aq0 = load_a(qrow, 0,  hl);   // d 0..31
    const v16bf aq1 = load_a(qrow, 32, hl);   // d 32..63

#if USE_TDM
    const uint32_t kst_off = (uint32_t)(uintptr_t)(void*)&Kst[w][0][0];
    const uint32_t vst_off = (uint32_t)(uintptr_t)(void*)&Vst[w][0][0];
#endif

    const float SCALE = 0.125f;               // 1/sqrt(64)
    float m_run[8], l_run[8], alpha[8];
    v8f O[4];
    #pragma unroll
    for (int r = 0; r < 8; ++r) { m_run[r] = -1e30f; l_run[r] = 0.f; }
    #pragma unroll
    for (int f = 0; f < 4; ++f) { v8f z = {}; O[f] = z; }

    const int qend = q0 + 15;
    for (int j = 0; j <= qend; j += 32) {
#if USE_TDM
        // Stage K tile (32 keys x 64 d) and V^T tile (64 d x 32 keys) via TDM.
        asm volatile("s_wait_dscnt 0" ::: "memory");  // LDS reads of prev chunk done
        tdm_load_2d(kst_off, Kh + (size_t)j * HD, HD, TT, HD, 32, HD);
        tdm_load_2d(vst_off, Vh + j,              TT, HD, 32, HD, TT);
        __builtin_amdgcn_s_wait_tensorcnt((short)0);
        asm volatile("" ::: "memory");
        const __bf16* kr0 = &Kst[w][l16][0];
        const __bf16* kr1 = &Kst[w][16 + l16][0];
#else
        const __bf16* kr0 = Kh + (size_t)(j + l16) * HD;
        const __bf16* kr1 = Kh + (size_t)(j + 16 + l16) * HD;
#endif
        // ---- S = Q * K^T for 32 keys (two 16-key subtiles) ----
        v8f c0 = {}, c1 = {};
        c0 = wmma_bf16(aq0, *reinterpret_cast<const v16bf*>(kr0 + hl * 16),      c0);
        c0 = wmma_bf16(aq1, *reinterpret_cast<const v16bf*>(kr0 + 32 + hl * 16), c0);
        c1 = wmma_bf16(aq0, *reinterpret_cast<const v16bf*>(kr1 + hl * 16),      c1);
        c1 = wmma_bf16(aq1, *reinterpret_cast<const v16bf*>(kr1 + 32 + hl * 16), c1);

        // ---- online softmax (rows M = r + 8*hl, col N = l16) ----
        #pragma unroll
        for (int r = 0; r < 8; ++r) {
            const int mloc = r + 8 * hl;
            const int qg = q0 + mloc;
            float s0 = c0[r] * SCALE;
            float s1 = c1[r] * SCALE;
            if (j + l16 > qg)      s0 = -1e30f;
            if (j + 16 + l16 > qg) s1 = -1e30f;
            float mx = rmax16(fmaxf(s0, s1));
            float mnew = fmaxf(m_run[r], mx);
            float a  = __expf(m_run[r] - mnew);
            float p0 = __expf(s0 - mnew);
            float p1 = __expf(s1 - mnew);
            float rs = rsum16(p0 + p1);
            l_run[r] = l_run[r] * a + rs;
            m_run[r] = mnew;
            alpha[r] = a;
            plds[w][mloc][l16]      = tobf(p0);
            plds[w][mloc][16 + l16] = tobf(p1);
        }
        #pragma unroll
        for (int f = 0; f < 4; ++f)
            #pragma unroll
            for (int r = 0; r < 8; ++r) O[f][r] *= alpha[r];

        wave_fence();
        asm volatile("s_wait_dscnt 0" ::: "memory");

        // P tile (16x32) as A-fragment from LDS
        AF pf;
        pf.q[0] = *reinterpret_cast<const uint4*>(&plds[w][l16][hl * 16]);
        pf.q[1] = *reinterpret_cast<const uint4*>(&plds[w][l16][hl * 16 + 8]);
        const v16bf ap = pf.v;
        wave_fence();

        // ---- O += P * V ----
        #pragma unroll
        for (int f = 0; f < 4; ++f) {
#if USE_TDM
            const __bf16* vr = &Vst[w][f * 16 + l16][0];
            v16bf bv = *reinterpret_cast<const v16bf*>(vr + hl * 16);
#else
            const __bf16* vr = Vh + (size_t)(f * 16 + l16) * TT;
            v16bf bv = *reinterpret_cast<const v16bf*>(vr + j + hl * 16);
#endif
            O[f] = wmma_bf16(ap, bv, O[f]);
        }
    }

    // ---- normalize + store y (bf16, row-major [token][C]) ----
    #pragma unroll
    for (int r = 0; r < 8; ++r) {
        const int mloc = r + 8 * hl;
        const float inv = 1.0f / l_run[r];
        const size_t tok = (size_t)bb * TT + q0 + mloc;
        #pragma unroll
        for (int f = 0; f < 4; ++f) {
            const int ch = h * HD + f * 16 + l16;
            yb[tok * CC + ch] = tobf(O[f][r] * inv);
        }
    }
}

// ---------------------------------------------------------------------------
// Kernel 3: output projection (4096 x 1024; 64x32 per wave) -> f32 out
// ---------------------------------------------------------------------------
__global__ __launch_bounds__(128)
void proj_kernel(const __bf16* __restrict__ yb,
                 const __bf16* __restrict__ WtP,
                 const float*  __restrict__ b_proj,
                 float* __restrict__ out)
{
    const int wid = blockIdx.x * 4 + (threadIdx.x >> 5);   // < 2048
    const int mb = wid >> 5, nb = wid & 31;
    const int m0 = mb * 64, n0 = nb * 32;

    v8f acc[4][2];
    #pragma unroll
    for (int i = 0; i < 4; ++i)
        #pragma unroll
        for (int j = 0; j < 2; ++j) { v8f z = {}; acc[i][j] = z; }

    mma_block42(yb, WtP, m0, n0, CC, acc);

    const int lane = threadIdx.x & 31;
    const int hl = lane >> 4, l16 = lane & 15;
    #pragma unroll
    for (int j = 0; j < 2; ++j) {
        const int n = n0 + 16 * j + l16;
        const float bv = b_proj[n];
        #pragma unroll
        for (int i = 0; i < 4; ++i)
            #pragma unroll
            for (int r = 0; r < 8; ++r) {
                const int m = m0 + 16 * i + r + 8 * hl;
                out[(size_t)m * CC + n] = acc[i][j][r] + bv;
            }
    }
}

// ---------------------------------------------------------------------------
// Launch: convert -> QKV GEMM -> flash attention -> projection (one stream)
// ---------------------------------------------------------------------------
extern "C" void kernel_launch(void* const* d_in, const int* in_sizes, int n_in,
                              void* d_out, int out_size, void* d_ws, size_t ws_size,
                              hipStream_t stream) {
    const float* x      = (const float*)d_in[0];
    // d_in[1] = mask (bool) — causality hardcoded
    const float* W_attn = (const float*)d_in[2];
    const float* b_attn = (const float*)d_in[3];
    const float* W_proj = (const float*)d_in[4];
    const float* b_proj = (const float*)d_in[5];

    char* ws = (char*)d_ws;
    size_t off = 0;
    __bf16* xb  = (__bf16*)(ws + off); off += (size_t)MTOT * CC * 2;
    __bf16* WtA = (__bf16*)(ws + off); off += (size_t)N3C * CC * 2;
    __bf16* WtP = (__bf16*)(ws + off); off += (size_t)CC * CC * 2;
    __bf16* Qb  = (__bf16*)(ws + off); off += (size_t)MTOT * CC * 2;
    __bf16* Kb  = (__bf16*)(ws + off); off += (size_t)MTOT * CC * 2;
    __bf16* Vt  = (__bf16*)(ws + off); off += (size_t)MTOT * CC * 2;
    __bf16* yb  = (__bf16*)(ws + off); off += (size_t)MTOT * CC * 2;

    cvt_kernel<<<4096, 256, 0, stream>>>(x, W_attn, W_proj, xb, WtA, WtP);

    // 64 x 96 wave-blocks, 4 waves/block
    qkv_kernel<<<(64 * 96) / 4, 128, 0, stream>>>(xb, WtA, b_attn, Qb, Kb, Vt);

    // B*H*(T/16) = 4096 waves, 4 waves/block
    attn_kernel<<<4096 / 4, 128, 0, stream>>>(Qb, Kb, Vt, yb);

    // 64 x 32 wave-blocks, 4 waves/block
    proj_kernel<<<(64 * 32) / 4, 128, 0, stream>>>(yb, WtP, b_proj, (float*)d_out);
}
